// SumformerModel_52501680226689
// MI455X (gfx1250) — matrix operand order repaired and checked by reading, not compiled, not run on hardware
//
#include <hip/hip_runtime.h>

#define NNODES 200000
#define GG     512
#define DD     256
#define OUTD   128
#define LAYERS 4
#define TR     64          // node rows per block (200000 % 64 == 0 -> 3125 blocks)
#define AL     (DD + 8)    // padded bf16 row stride (528B: rotates LDS banks)
#define CL     (2*DD + 8)  // padded bf16 row stride for the 512-wide concat (1040B)
#define EPSF   1e-5f

typedef __bf16 bf16;
typedef __attribute__((ext_vector_type(16))) __bf16 v16bf;
typedef __attribute__((ext_vector_type(8)))  __bf16 v8bf;
typedef __attribute__((ext_vector_type(4)))  __bf16 v4bf;
typedef __attribute__((ext_vector_type(8)))  float  v8f;

// ---------------- CDNA5 async global->LDS (per-lane 16B, ASYNCcnt-tracked) ----------------
// generic LDS pointer low 32 bits == LDS byte address
static __device__ __forceinline__ void async_ld16(unsigned lds_off, const void* g) {
  asm volatile("global_load_async_to_lds_b128 %0, %1, off"
               :: "v"(lds_off), "v"((unsigned long long)(size_t)g)
               : "memory");
}
static __device__ __forceinline__ void wait_async0() {
  asm volatile("s_wait_asynccnt 0x0" ::: "memory");
}

// stage one 64-row bf16 tile (row stride 256 bf16 in global, dld bf16 in LDS):
// wave w copies rows [w*8, w*8+8); lane copies 16B of the 512B row.
static __device__ __forceinline__ void async_stage_tile(const bf16* gsrc, unsigned lds_base,
                                                        int dld_bytes, int wv, int lane) {
#pragma unroll
  for (int rr = 0; rr < 8; ++rr) {
    int row = wv * 8 + rr;
    async_ld16(lds_base + row * dld_bytes + lane * 16,
               gsrc + (size_t)row * DD + lane * 8);
  }
  wait_async0();
}

// ---------------- WMMA fragment helpers (wave32, 16x16x32 bf16) ----------------
static __device__ __forceinline__ v16bf lds_a_frag(const bf16* base, int ld,
                                                   int mt, int kt, int lane) {
  const bf16* p = base + (mt * 16 + (lane & 15)) * ld + kt * 32 + ((lane >> 4) << 3);
  v8bf lo = *(const v8bf*)p;
  v8bf hi = *(const v8bf*)(p + 16);
  v16bf a;
#pragma unroll
  for (int i = 0; i < 8; ++i) { a[i] = lo[i]; a[i + 8] = hi[i]; }
  return a;
}

static __device__ __forceinline__ v16bf glb_b_frag(const bf16* packed, int frag, int lane) {
  return *(const v16bf*)(packed + ((size_t)frag * 32 + lane) * 16);
}

// register-blocked GEMM: one wave -> 8 C tiles (same mt, 8 consecutive nt).
// A frag loaded once per kt and fed to 8 independent WMMAs (8 live accumulators).
// NOTE: B loads deliberately left inside the j-loop for the compiler to schedule;
// manual prefetch rotation provoked accumulator copies + hazard NOPs.
template <int KT, int NTILES>
static __device__ __forceinline__ void gemm_row(const bf16* a_lds, int ald,
                                                const bf16* bp, v8f acc[8],
                                                int mt, int ntb, int lane) {
#pragma unroll
  for (int j = 0; j < 8; ++j) acc[j] = (v8f){};
  for (int kt = 0; kt < KT; ++kt) {
    v16bf a = lds_a_frag(a_lds, ald, mt, kt, lane);
#pragma unroll
    for (int j = 0; j < 8; ++j) {
      v16bf b = glb_b_frag(bp, kt * NTILES + ntb + j, lane);
      acc[j] = __builtin_amdgcn_wmma_f32_16x16x32_bf16(false, a, false, b, (short)0, acc[j],
                                                       false, false);
    }
  }
}

// ---------------- LayerNorm helpers (per 64x256 f32 tile in LDS) ----------------
static __device__ __forceinline__ void ln_stats(const float* hs, float* psum, float* psum2,
                                                float* mu_s, float* ri_s, int tid) {
  int r = tid >> 2, q = tid & 3;
  float s = 0.f, ss = 0.f;
  const float* hp = hs + r * DD + q * 64;
#pragma unroll 8
  for (int c = 0; c < 64; ++c) { float v = hp[c]; s += v; ss += v * v; }
  psum[tid] = s; psum2[tid] = ss;
  __syncthreads();
  if (tid < TR) {
    float S  = psum[tid*4] + psum[tid*4+1] + psum[tid*4+2] + psum[tid*4+3];
    float SS = psum2[tid*4] + psum2[tid*4+1] + psum2[tid*4+2] + psum2[tid*4+3];
    float mu = S * (1.f / DD);
    float var = SS * (1.f / DD) - mu * mu;
    mu_s[tid] = mu;
    ri_s[tid] = rsqrtf(var + EPSF);
  }
  __syncthreads();
}

static __device__ __forceinline__ void ln_apply_bf16(const float* hs, bf16* dst, int dld,
                                                     const float* g, const float* b,
                                                     const float* mu_s, const float* ri_s, int tid) {
  for (int i = tid; i < TR * DD / 4; i += 256) {
    int r = i >> 6, c4 = (i & 63) * 4;
    float4 v  = *(const float4*)(hs + r * DD + c4);
    float4 gg = *(const float4*)(g + c4);
    float4 bb = *(const float4*)(b + c4);
    float mu = mu_s[r], ri = ri_s[r];
    v4bf o = { (bf16)((v.x - mu) * ri * gg.x + bb.x),
               (bf16)((v.y - mu) * ri * gg.y + bb.y),
               (bf16)((v.z - mu) * ri * gg.z + bb.z),
               (bf16)((v.w - mu) * ri * gg.w + bb.w) };
    *(v4bf*)(dst + r * dld + c4) = o;
  }
}

// out_ln: write both f32 x (next-layer residual/LN math) and bf16 mirror (GEMM-A staging)
static __device__ __forceinline__ void ln_apply_out(const float* hs, float* dstf, bf16* dsth,
                                                    const float* g, const float* b,
                                                    const float* mu_s, const float* ri_s, int tid) {
  for (int i = tid; i < TR * DD / 4; i += 256) {
    int r = i >> 6, c4 = (i & 63) * 4;
    float4 v  = *(const float4*)(hs + r * DD + c4);
    float4 gg = *(const float4*)(g + c4);
    float4 bb = *(const float4*)(b + c4);
    float mu = mu_s[r], ri = ri_s[r];
    float4 o;
    o.x = (v.x - mu) * ri * gg.x + bb.x;
    o.y = (v.y - mu) * ri * gg.y + bb.y;
    o.z = (v.z - mu) * ri * gg.z + bb.z;
    o.w = (v.w - mu) * ri * gg.w + bb.w;
    *(float4*)(dstf + (size_t)r * DD + c4) = o;
    v4bf h = { (bf16)o.x, (bf16)o.y, (bf16)o.z, (bf16)o.w };
    *(v4bf*)(dsth + (size_t)r * DD + c4) = h;
  }
}

// ---------------- weight packing: f32 [K,N] row-major -> bf16 B-fragments ----------------
__global__ void pack_b_kernel(const float* __restrict__ W, bf16* __restrict__ P,
                              int Kdim, int Ndim) {
  int total = Kdim * Ndim;
  int ntiles = Ndim >> 4;
  for (int tid = blockIdx.x * blockDim.x + threadIdx.x; tid < total;
       tid += gridDim.x * blockDim.x) {
    int h = tid & 15;
    int l = (tid >> 4) & 31;
    int f = tid >> 9;
    int nt = f % ntiles;
    int kt = f / ntiles;
    int kk = ((h >> 3) << 4) + ((l >> 4) << 3) + (h & 7);
    int nn = l & 15;
    P[tid] = (bf16)W[(kt * 32 + kk) * Ndim + nt * 16 + nn];
  }
}

// ---------------- one-time x f32 -> bf16 mirror ----------------
__global__ void cvt_bf16_kernel(const float* __restrict__ src, bf16* __restrict__ dst, int n4) {
  for (int i = blockIdx.x * blockDim.x + threadIdx.x; i < n4; i += gridDim.x * blockDim.x) {
    float4 v = ((const float4*)src)[i];
    v4bf o = { (bf16)v.x, (bf16)v.y, (bf16)v.z, (bf16)v.w };
    *(v4bf*)(dst + (size_t)i * 4) = o;
  }
}

// ---------------- phi MLP + pooled segmented scatter-add ----------------
__global__ __launch_bounds__(256) void phi_kernel(
    const bf16* __restrict__ xb, const int* __restrict__ batch,
    const bf16* __restrict__ w1p, const float* __restrict__ b1,
    const float* __restrict__ lng, const float* __restrict__ lnb,
    const bf16* __restrict__ w2p, const float* __restrict__ b2,
    float* __restrict__ gsum) {
  extern __shared__ unsigned char smem[];
  bf16*  xa = (bf16*)smem;                         // 64 x AL bf16 (GEMM A)
  float* hs = (float*)(smem + TR * AL * 2);        // 64 x 256 f32
  __shared__ int   batch_s[TR];
  __shared__ float psum[256], psum2[256], mu_s[TR], ri_s[TR];

  const int tid = threadIdx.x;
  const int lane = tid & 31, wv = tid >> 5;
  const int row0 = blockIdx.x * TR;

  // async-stage bf16 x tile straight into LDS (no VGPR round-trip)
  async_stage_tile(xb + (size_t)row0 * DD, (unsigned)(size_t)xa, AL * 2, wv, lane);
  if (tid < TR) batch_s[tid] = batch[row0 + tid];
  __syncthreads();

  const int mt = wv >> 1, ntb = (wv & 1) * 8;
  const int mbase = mt * 16 + ((lane >> 4) << 3);
  const int nlo = lane & 15;
  v8f acc[8];

  // GEMM1: h1 = relu(x @ w1 + b1) -> hs
  gemm_row<8, 16>(xa, AL, w1p, acc, mt, ntb, lane);
#pragma unroll
  for (int j = 0; j < 8; ++j) {
    int nn = (ntb + j) * 16 + nlo;
    float bias = b1[nn];
#pragma unroll
    for (int r = 0; r < 8; ++r) {
      float v = acc[j][r] + bias;
      hs[(mbase + r) * DD + nn] = v > 0.f ? v : 0.f;
    }
  }
  __syncthreads();

  // LayerNorm(h1) -> xa (bf16, reuse)
  ln_stats(hs, psum, psum2, mu_s, ri_s, tid);
  ln_apply_bf16(hs, xa, AL, lng, lnb, mu_s, ri_s, tid);
  __syncthreads();

  // GEMM2: phi = h1n @ w2 + b2 -> hs
  gemm_row<8, 16>(xa, AL, w2p, acc, mt, ntb, lane);
#pragma unroll
  for (int j = 0; j < 8; ++j) {
    int nn = (ntb + j) * 16 + nlo;
    float bias = b2[nn];
#pragma unroll
    for (int r = 0; r < 8; ++r)
      hs[(mbase + r) * DD + nn] = acc[j][r] + bias;
  }
  __syncthreads();

  // block-local segmented reduction (batch sorted) -> few f32 atomics per graph
  {
    int col = tid;
    float a = 0.f;
    int cur = batch_s[0];
    for (int r = 0; r < TR; ++r) {
      int b = batch_s[r];
      if (b != cur) { atomicAdd(&gsum[cur * DD + col], a); a = 0.f; cur = b; }
      a += hs[r * DD + col];
    }
    atomicAdd(&gsum[cur * DD + col], a);
  }
}

// ---------------- psi MLP on concat(x, gsum[batch]) + residual + out_ln (in place) ----------------
__global__ __launch_bounds__(256) void psi_kernel(
    float* __restrict__ xg, bf16* __restrict__ xb, const int* __restrict__ batch,
    const float* __restrict__ gsum,
    const bf16* __restrict__ w1p, const float* __restrict__ b1,
    const float* __restrict__ lng, const float* __restrict__ lnb,
    const bf16* __restrict__ w2p, const float* __restrict__ b2,
    const float* __restrict__ og, const float* __restrict__ ob) {
  extern __shared__ unsigned char smem[];
  bf16*  ca = (bf16*)smem;                          // 64 x CL bf16 (concat A)
  float* hs = (float*)(smem + TR * CL * 2);         // 64 x 256 f32
  bf16*  hb = (bf16*)smem;                          // 64 x AL bf16, aliases ca (dead after GEMM1)
  __shared__ int   batch_s[TR];
  __shared__ float psum[256], psum2[256], mu_s[TR], ri_s[TR];

  const int tid = threadIdx.x;
  const int lane = tid & 31, wv = tid >> 5;
  const int row0 = blockIdx.x * TR;

  if (tid < TR) batch_s[tid] = batch[row0 + tid];

  // async-stage x half of the concat (bf16 mirror), gsum half converts via VGPRs
  async_stage_tile(xb + (size_t)row0 * DD, (unsigned)(size_t)ca, CL * 2, wv, lane);
  __syncthreads();   // batch_s visible
  for (int i = tid; i < TR * DD / 4; i += 256) {
    int r = i >> 6, c4 = (i & 63) * 4;
    float4 v = *(const float4*)(gsum + (size_t)batch_s[r] * DD + c4);
    v4bf o = { (bf16)v.x, (bf16)v.y, (bf16)v.z, (bf16)v.w };
    *(v4bf*)(ca + r * CL + DD + c4) = o;
  }
  __syncthreads();

  const int mt = wv >> 1, ntb = (wv & 1) * 8;
  const int mbase = mt * 16 + ((lane >> 4) << 3);
  const int nlo = lane & 15;
  v8f acc[8];

  // GEMM1 (K=512): h1 = relu(concat @ psi_w1 + b1) -> hs
  gemm_row<16, 16>(ca, CL, w1p, acc, mt, ntb, lane);
#pragma unroll
  for (int j = 0; j < 8; ++j) {
    int nn = (ntb + j) * 16 + nlo;
    float bias = b1[nn];
#pragma unroll
    for (int r = 0; r < 8; ++r) {
      float v = acc[j][r] + bias;
      hs[(mbase + r) * DD + nn] = v > 0.f ? v : 0.f;
    }
  }
  __syncthreads();

  // psi LayerNorm -> hb (bf16)
  ln_stats(hs, psum, psum2, mu_s, ri_s, tid);
  ln_apply_bf16(hs, hb, AL, lng, lnb, mu_s, ri_s, tid);
  __syncthreads();

  // GEMM2 (K=256): h2 = h1n @ psi_w2 + b2 -> hs
  gemm_row<8, 16>(hb, AL, w2p, acc, mt, ntb, lane);
#pragma unroll
  for (int j = 0; j < 8; ++j) {
    int nn = (ntb + j) * 16 + nlo;
    float bias = b2[nn];
#pragma unroll
    for (int r = 0; r < 8; ++r)
      hs[(mbase + r) * DD + nn] = acc[j][r] + bias;
  }
  __syncthreads();

  // residual: hs += x (f32, coalesced float4)
  for (int i = tid; i < TR * DD / 4; i += 256) {
    int r = i >> 6, c4 = (i & 63) * 4;
    float4 xv = *(const float4*)(xg + (size_t)(row0 + r) * DD + c4);
    float4 hv = *(const float4*)(hs + r * DD + c4);
    hv.x += xv.x; hv.y += xv.y; hv.z += xv.z; hv.w += xv.w;
    *(float4*)(hs + r * DD + c4) = hv;
  }
  __syncthreads();

  // out_ln -> write new x (f32 in place + bf16 mirror)
  ln_stats(hs, psum, psum2, mu_s, ri_s, tid);
  ln_apply_out(hs, xg + (size_t)row0 * DD, xb + (size_t)row0 * DD, og, ob, mu_s, ri_s, tid);
}

// ---------------- head: out = x @ head_w + head_b ----------------
__global__ __launch_bounds__(256) void head_kernel(
    const bf16* __restrict__ xb, const bf16* __restrict__ wp,
    const float* __restrict__ hbias, float* __restrict__ out) {
  extern __shared__ unsigned char smem[];
  bf16* xa = (bf16*)smem;                           // 64 x AL bf16

  const int tid = threadIdx.x;
  const int lane = tid & 31, wv = tid >> 5;
  const int row0 = blockIdx.x * TR;

  async_stage_tile(xb + (size_t)row0 * DD, (unsigned)(size_t)xa, AL * 2, wv, lane);
  __syncthreads();

  // 32 tiles (4 mt x 8 nt) over 8 waves -> each wave: same mt, 4 nt, reg-blocked
  const int mt = wv >> 1, ntb = (wv & 1) * 4;
  const int mbase = mt * 16 + ((lane >> 4) << 3);
  const int nlo = lane & 15;

  v8f acc[4];
#pragma unroll
  for (int j = 0; j < 4; ++j) acc[j] = (v8f){};
  for (int kt = 0; kt < 8; ++kt) {
    v16bf a = lds_a_frag(xa, AL, mt, kt, lane);
#pragma unroll
    for (int j = 0; j < 4; ++j) {
      v16bf b = glb_b_frag(wp, kt * 8 + ntb + j, lane);
      acc[j] = __builtin_amdgcn_wmma_f32_16x16x32_bf16(false, a, false, b, (short)0, acc[j],
                                                       false, false);
    }
  }
#pragma unroll
  for (int j = 0; j < 4; ++j) {
    int nn = (ntb + j) * 16 + nlo;
    float bias = hbias[nn];
#pragma unroll
    for (int r = 0; r < 8; ++r)
      out[(size_t)(row0 + mbase + r) * OUTD + nn] = acc[j][r] + bias;
  }
}

// ---------------- host launcher ----------------
extern "C" void kernel_launch(void* const* d_in, const int* in_sizes, int n_in,
                              void* d_out, int out_size, void* d_ws, size_t ws_size,
                              hipStream_t stream) {
  (void)in_sizes; (void)n_in; (void)out_size; (void)ws_size;
  const float* x_in   = (const float*)d_in[0];
  const int*   batch  = (const int*)d_in[1];
  const float* phi_w1 = (const float*)d_in[2];
  const float* phi_b1 = (const float*)d_in[3];
  const float* phi_lg = (const float*)d_in[4];
  const float* phi_lb = (const float*)d_in[5];
  const float* phi_w2 = (const float*)d_in[6];
  const float* phi_b2 = (const float*)d_in[7];
  const float* psi_w1 = (const float*)d_in[8];
  const float* psi_b1 = (const float*)d_in[9];
  const float* psi_lg = (const float*)d_in[10];
  const float* psi_lb = (const float*)d_in[11];
  const float* psi_w2 = (const float*)d_in[12];
  const float* psi_b2 = (const float*)d_in[13];
  const float* out_lg = (const float*)d_in[14];
  const float* out_lb = (const float*)d_in[15];
  const float* head_w = (const float*)d_in[16];
  const float* head_b = (const float*)d_in[17];

  unsigned char* ws = (unsigned char*)d_ws;
  size_t off = 0;
  float* x_buf = (float*)(ws + off);  off += (size_t)NNODES * DD * 4;
  bf16*  xb    = (bf16*)(ws + off);   off += (size_t)NNODES * DD * 2;
  float* gsum  = (float*)(ws + off);  off += (size_t)GG * DD * 4;
  bf16*  pw1   = (bf16*)(ws + off);   off += (size_t)LAYERS * DD * DD * 2;
  bf16*  pw2   = (bf16*)(ws + off);   off += (size_t)LAYERS * DD * DD * 2;
  bf16*  qw1   = (bf16*)(ws + off);   off += (size_t)LAYERS * 2 * DD * DD * 2;
  bf16*  qw2   = (bf16*)(ws + off);   off += (size_t)LAYERS * DD * DD * 2;
  bf16*  hw    = (bf16*)(ws + off);   off += (size_t)DD * OUTD * 2;

  static bool attr_done = false;  // host-side only; deterministic
  if (!attr_done) {
    hipFuncSetAttribute((const void*)phi_kernel, hipFuncAttributeMaxDynamicSharedMemorySize,
                        TR * AL * 2 + TR * DD * 4);
    hipFuncSetAttribute((const void*)psi_kernel, hipFuncAttributeMaxDynamicSharedMemorySize,
                        TR * CL * 2 + TR * DD * 4);
    hipFuncSetAttribute((const void*)head_kernel, hipFuncAttributeMaxDynamicSharedMemorySize,
                        TR * AL * 2);
    attr_done = true;
  }

  // pack all weights to bf16 B-fragment layout (L2-resident afterwards)
  for (int i = 0; i < LAYERS; ++i) {
    const int t1 = DD * DD;
    const int t2 = 2 * DD * DD;
    pack_b_kernel<<<(t1 + 255) / 256, 256, 0, stream>>>(phi_w1 + (size_t)i * t1, pw1 + (size_t)i * t1, DD, DD);
    pack_b_kernel<<<(t1 + 255) / 256, 256, 0, stream>>>(phi_w2 + (size_t)i * t1, pw2 + (size_t)i * t1, DD, DD);
    pack_b_kernel<<<(t2 + 255) / 256, 256, 0, stream>>>(psi_w1 + (size_t)i * t2, qw1 + (size_t)i * t2, 2 * DD, DD);
    pack_b_kernel<<<(t1 + 255) / 256, 256, 0, stream>>>(psi_w2 + (size_t)i * t1, qw2 + (size_t)i * t1, DD, DD);
  }
  pack_b_kernel<<<(DD * OUTD + 255) / 256, 256, 0, stream>>>(head_w, hw, DD, OUTD);

  hipMemcpyAsync(x_buf, x_in, (size_t)NNODES * DD * 4, hipMemcpyDeviceToDevice, stream);
  cvt_bf16_kernel<<<2048, 256, 0, stream>>>(x_in, xb, NNODES * DD / 4);

  const int nblocks = NNODES / TR;
  for (int i = 0; i < LAYERS; ++i) {
    hipMemsetAsync(gsum, 0, (size_t)GG * DD * 4, stream);
    phi_kernel<<<nblocks, 256, TR * AL * 2 + TR * DD * 4, stream>>>(
        xb, batch,
        pw1 + (size_t)i * DD * DD, phi_b1 + (size_t)i * DD,
        phi_lg + (size_t)i * DD,   phi_lb + (size_t)i * DD,
        pw2 + (size_t)i * DD * DD, phi_b2 + (size_t)i * DD, gsum);
    psi_kernel<<<nblocks, 256, TR * CL * 2 + TR * DD * 4, stream>>>(
        x_buf, xb, batch, gsum,
        qw1 + (size_t)i * 2 * DD * DD, psi_b1 + (size_t)i * DD,
        psi_lg + (size_t)i * DD,       psi_lb + (size_t)i * DD,
        qw2 + (size_t)i * DD * DD,     psi_b2 + (size_t)i * DD,
        out_lg + (size_t)i * DD,       out_lb + (size_t)i * DD);
  }
  head_kernel<<<nblocks, 256, TR * AL * 2, stream>>>(xb, hw, head_b, (float*)d_out);
}
